// EnhancedMamba2DBlock_24438363914546
// MI455X (gfx1250) — matrix operand order repaired and checked by reading, not compiled, not run on hardware
//
#include <hip/hip_runtime.h>
#include <hip/hip_bf16.h>
#include <math.h>

// ---------------------------------------------------------------------------
// EnhancedMamba2DBlock forward for MI455X (gfx1250, wave32, WMMA).
//
// Pipeline (all intermediates token-major NHWC in d_ws, fp32 unless noted):
//   prep:  f32->f16 weight conversion (5 mats) + BN affine folding (5 sets)
//   A:  LN (global) + BN_in (local) + NCHW->NHWC transpose
//   B:  uv   = xln @ ex_w.T + ex_b            (WMMA, K=128, N=512)
//   C:  vs   = silu(dw3x3(v))                 (depthwise, NHWC)
//   D:  xdbl = vs @ x_proj.T                  (WMMA, K=256, N=48)
//   E:  dt   = softplus(xdbl[:,:16]@dt_w.T+b) (VALU, K=16)
//   F:  scan -> vcomb = y*silu(u) + vs*rsilu(u)  (wave32 scan, shfl B/C)
//   G:  gp   = vcomb @ proj_w.T + proj_b      (WMMA, K=256, N=128)
//   H:  l1s  = silu(bn1(xbn @ pw_ex.T))       (WMMA, K=128, N=256)
//   I:  l2   = silu(bn2(dw3x3(l1s)))
//   J:  lp   = bn3(l2 @ pw_pr.T)              (WMMA, K=256, N=128)
//   K/L: gates (32-wide MLP, wave shuffle LN)
//   M:  out = x + bn_norm(gp*gate_l(lp) + lp*gate_g(gp))  (NHWC->NCHW)
// ---------------------------------------------------------------------------

typedef __attribute__((ext_vector_type(16))) _Float16 v16h;
typedef __attribute__((ext_vector_type(8)))  _Float16 v8h;
typedef __attribute__((ext_vector_type(8)))  float    v8f;

#define NB   8
#define NCc  128
#define NHh  64
#define NWw  64
#define NCE  256
#define NLl  4096            // H*W
#define NT   32768           // B*H*W tokens

__device__ __forceinline__ float sigm_(float x) { return 1.0f / (1.0f + __expf(-x)); }
__device__ __forceinline__ float silu_(float x) { return x * sigm_(x); }

// ---------------------------------------------------------------- weight prep
__global__ void cvt_f16_kernel(const float* __restrict__ in, _Float16* __restrict__ out, int n)
{
    int i = blockIdx.x * 256 + threadIdx.x;
    if (i < n) out[i] = (_Float16)in[i];
}

__global__ void bn_affine_kernel(const float* __restrict__ g, const float* __restrict__ b,
                                 const float* __restrict__ m, const float* __restrict__ v,
                                 float* __restrict__ s, float* __restrict__ sh, int n)
{
    int i = blockIdx.x * 256 + threadIdx.x;
    if (i < n) {
        float sc = g[i] * rsqrtf(v[i] + 1e-5f);
        s[i]  = sc;
        sh[i] = b[i] - m[i] * sc;
    }
}

// ------------------------------------------- LN + BN_in + NCHW->NHWC transpose
__global__ __launch_bounds__(256)
void lnbn_transpose_kernel(const float* __restrict__ x,
                           const float* __restrict__ lng, const float* __restrict__ lnb,
                           const float* __restrict__ bs,  const float* __restrict__ bsh,
                           float* __restrict__ xln, float* __restrict__ xbn)
{
    __shared__ float tile[64 * 132];
    __shared__ float psum[256];
    __shared__ float psq[256];
    __shared__ float murs[64 * 2];

    int bh = blockIdx.x;                 // 0..511 : (b,h)
    int b  = bh >> 6, h = bh & 63;
    int tid = threadIdx.x;
    int w  = tid & 63, cc = tid >> 6;    // cc in 0..3

    const float* xb = x + ((size_t)b * NCc) * NLl + h * 64 + w;
    float s = 0.f, sq = 0.f;
    #pragma unroll 4
    for (int i = 0; i < 32; ++i) {
        int c = cc + i * 4;
        float v = xb[(size_t)c * NLl];
        tile[w * 132 + c] = v;
        s += v; sq += v * v;
    }
    psum[tid] = s; psq[tid] = sq;
    __syncthreads();

    if (tid < 64) {
        float ss = 0.f, qq = 0.f;
        #pragma unroll
        for (int i = 0; i < 4; ++i) { ss += psum[tid + i * 64]; qq += psq[tid + i * 64]; }
        float mu  = ss * (1.0f / 128.0f);
        float var = qq * (1.0f / 128.0f) - mu * mu;
        murs[tid * 2]     = mu;
        murs[tid * 2 + 1] = rsqrtf(var + 1e-6f);   // NORM_EPS
    }
    __syncthreads();

    int w2 = tid >> 2, q = tid & 3;
    float mu = murs[w2 * 2], rs = murs[w2 * 2 + 1];
    size_t t = ((size_t)b << 12) + (h << 6) + w2;
    float* oln = xln + t * NCc + q * 32;
    float* obn = xbn + t * NCc + q * 32;
    #pragma unroll 4
    for (int j = 0; j < 32; ++j) {
        int c = q * 32 + j;
        float v = tile[w2 * 132 + c];
        oln[j] = (v - mu) * rs * lng[c] + lnb[c];
        obn[j] = v * bs[c] + bsh[c];
    }
}

// --------------------------------------------------------------- WMMA GEMM
// Out(T,N) = epilogue( A(T,K=KF*32) @ W(N,K)^T )
// AFF: 0 = identity, 1 = +shift, 2 = *scale+shift.  ACT: 0 none, 1 silu.
template<int KF, int ACT, int AFF>
__global__ __launch_bounds__(128)
void wmma_gemm_kernel(const float* __restrict__ A, const _Float16* __restrict__ W,
                      float* __restrict__ Out, int N,
                      const float* __restrict__ scale, const float* __restrict__ shift)
{
    const int K = KF * 32;
    int wave = blockIdx.x * 4 + (threadIdx.x >> 5);
    int lane = threadIdx.x & 31;
    int m0   = wave * 16;
    int row  = lane & 15;
    int kb   = (lane >> 4) * 8;    // A frag K sub-offset
    int kb2  = (lane >> 4) * 8;    // B frag K sub-offset

    // Load this wave's 16xK A tile into f16 fragments (kept in VGPRs for all N tiles)
    v16h afrag[KF];
    const float* arow = A + (size_t)(m0 + row) * K;
    #pragma unroll
    for (int f = 0; f < KF; ++f) {
        const float* p = arow + f * 32;
        float4 a0 = *(const float4*)(p + kb);
        float4 a1 = *(const float4*)(p + kb + 4);
        float4 a2 = *(const float4*)(p + 16 + kb);
        float4 a3 = *(const float4*)(p + 16 + kb + 4);
        v16h a;
        a[0] = (_Float16)a0.x; a[1] = (_Float16)a0.y; a[2]  = (_Float16)a0.z; a[3]  = (_Float16)a0.w;
        a[4] = (_Float16)a1.x; a[5] = (_Float16)a1.y; a[6]  = (_Float16)a1.z; a[7]  = (_Float16)a1.w;
        a[8] = (_Float16)a2.x; a[9] = (_Float16)a2.y; a[10] = (_Float16)a2.z; a[11] = (_Float16)a2.w;
        a[12]= (_Float16)a3.x; a[13]= (_Float16)a3.y; a[14] = (_Float16)a3.z; a[15] = (_Float16)a3.w;
        afrag[f] = a;
    }

    int col = lane & 15;
    for (int n0 = 0; n0 < N; n0 += 16) {
        // batch all weight-fragment loads for this N tile (one clause, one wait)
        const v8h* wp = (const v8h*)(W + (size_t)(n0 + col) * K + kb2);
        v8h wl[KF * 2];
        #pragma unroll
        for (int f = 0; f < KF; ++f) {
            wl[2 * f]     = wp[f * 4];
            wl[2 * f + 1] = wp[f * 4 + 2];
        }
        v8f acc = {};
        #pragma unroll
        for (int f = 0; f < KF; ++f) {
            v16h bfrag;
            #pragma unroll
            for (int j = 0; j < 8; ++j) { bfrag[j] = wl[2 * f][j]; bfrag[8 + j] = wl[2 * f + 1][j]; }
            acc = __builtin_amdgcn_wmma_f32_16x16x32_f16(
                false, afrag[f], false, bfrag, (short)0, acc, false, false);
        }
        int n = n0 + col;
        float sc = (AFF == 2) ? scale[n] : 1.0f;
        float sh = (AFF >= 1) ? shift[n] : 0.0f;
        float* orow = Out + (size_t)(m0 + ((lane >> 4) << 3)) * N + n;
        #pragma unroll
        for (int r = 0; r < 8; ++r) {
            float v = acc[r] * sc + sh;
            if (ACT == 1) v = silu_(v);
            orow[(size_t)r * N] = v;
        }
    }
}

// ------------------------------------------------- depthwise 3x3 (NHWC), silu
__global__ __launch_bounds__(256)
void dwconv_kernel(const float* __restrict__ in, int instride,
                   const float* __restrict__ w9,
                   const float* __restrict__ scale, const float* __restrict__ shift,
                   float* __restrict__ out)
{
    int gid = blockIdx.x * 256 + threadIdx.x;
    int t = gid >> 6;
    int c = (gid & 63) << 2;
    int b = t >> 12, hw = t & 4095, h = hw >> 6, w = hw & 63;

    float wk[4][9];
    #pragma unroll
    for (int i = 0; i < 4; ++i)
        #pragma unroll
        for (int k = 0; k < 9; ++k) wk[i][k] = w9[(c + i) * 9 + k];

    float ax = 0.f, ay = 0.f, az = 0.f, aw = 0.f;
    #pragma unroll
    for (int ky = 0; ky < 3; ++ky) {
        int hh = h + ky - 1;
        if ((unsigned)hh >= (unsigned)NHh) continue;
        #pragma unroll
        for (int kx = 0; kx < 3; ++kx) {
            int ww = w + kx - 1;
            if ((unsigned)ww >= (unsigned)NWw) continue;
            size_t tt = ((size_t)b << 12) + (hh << 6) + ww;
            float4 vv = *(const float4*)(in + tt * instride + c);
            int k = ky * 3 + kx;
            ax += vv.x * wk[0][k]; ay += vv.y * wk[1][k];
            az += vv.z * wk[2][k]; aw += vv.w * wk[3][k];
        }
    }
    float o4[4] = { ax, ay, az, aw };
    float* op = out + (size_t)t * NCE + c;
    #pragma unroll
    for (int i = 0; i < 4; ++i) {
        float v = o4[i];
        if (scale) v = v * scale[c + i] + shift[c + i];
        op[i] = silu_(v);
    }
}

// ------------------------------------------- dt = softplus(xdbl[:,:16]@dt_w.T+b)
__global__ __launch_bounds__(256)
void dtproj_kernel(const float* __restrict__ xdbl, const float* __restrict__ dtw,
                   const float* __restrict__ dtb, float* __restrict__ dtout)
{
    int gid = blockIdx.x * 256 + threadIdx.x;
    int t = gid >> 6;
    int c0 = (gid & 63) << 2;
    const float* r = xdbl + (size_t)t * 48;
    float xv[16];
    #pragma unroll
    for (int k = 0; k < 16; ++k) xv[k] = r[k];
    #pragma unroll
    for (int i = 0; i < 4; ++i) {
        int c = c0 + i;
        float acc = dtb[c];
        const float* wr = dtw + c * 16;
        #pragma unroll
        for (int k = 0; k < 16; ++k) acc += xv[k] * wr[k];
        float sp = (acc > 20.f) ? acc : log1pf(__expf(acc));
        dtout[(size_t)t * NCE + c] = sp;
    }
}

// ------------------------------------------------------- selective scan (S6)
// one lane per (b,ce); B/C broadcast via shuffles; fused gate/combine epilogue
__global__ __launch_bounds__(32)
void scan_kernel(const float* __restrict__ xdbl, const float* __restrict__ dt,
                 const float* __restrict__ vs,   const float* __restrict__ uv,
                 const float* __restrict__ A_log, const float* __restrict__ Dp,
                 float* __restrict__ vcomb)
{
    int b  = blockIdx.x >> 3;
    int cg = blockIdx.x & 7;
    int lane = threadIdx.x;
    int ce = cg * 32 + lane;

    float aA[16], h[16];
    #pragma unroll
    for (int n = 0; n < 16; ++n) {
        aA[n] = -__expf(A_log[ce * 16 + n]);   // A = -exp(A_log)
        h[n] = 0.f;
    }
    float Dce = Dp[ce];
    size_t tbase = (size_t)b << 12;

    for (int l = 0; l < NLl; ++l) {
        size_t t = tbase + l;
        float bc  = xdbl[t * 48 + 16 + lane];        // lanes 0-15: B, 16-31: C
        float dtv = dt[t * NCE + ce];
        float uvs = vs[t * NCE + ce];
        float uo  = uv[t * 512 + ce];                // u = first 256 cols of uv
        float du  = dtv * uvs;
        float y = 0.f;
        #pragma unroll
        for (int n = 0; n < 16; ++n) {
            float dA = __expf(dtv * aA[n]);          // exp(dt * A)
            float Bn = __shfl(bc, n, 32);
            float Cn = __shfl(bc, n + 16, 32);
            h[n] = h[n] * dA + du * Bn;
            y += h[n] * Cn;
        }
        y += uvs * Dce;
        float sg = sigm_(uo);
        vcomb[t * NCE + ce] = y * (uo * sg) + uvs * (uo * (1.0f - sg));
    }
}

// ------------------------------------------------------------- gating MLP
// one wave per token: 128->32 (dot), LN(32) via shuffles, silu, 32->128, sigmoid
__global__ __launch_bounds__(256)
void gating_kernel(const float* __restrict__ in,
                   const float* __restrict__ w1, const float* __restrict__ lng,
                   const float* __restrict__ lnb, const float* __restrict__ w2,
                   const float* __restrict__ b2, float* __restrict__ out)
{
    __shared__ float hs[8][32];
    int wid = threadIdx.x >> 5, lane = threadIdx.x & 31;
    int t = blockIdx.x * 8 + wid;

    const float* rowp = in + (size_t)t * NCc;
    const float* wr = w1 + lane * NCc;
    float acc = 0.f;
    #pragma unroll 8
    for (int k = 0; k < NCc; k += 4) {
        float4 a = *(const float4*)(rowp + k);
        float4 b = *(const float4*)(wr + k);
        acc += a.x * b.x + a.y * b.y + a.z * b.z + a.w * b.w;
    }
    float s = acc;
    #pragma unroll
    for (int o = 16; o > 0; o >>= 1) s += __shfl_xor(s, o, 32);
    float mu = s * (1.0f / 32.0f);
    float d = acc - mu;
    float vq = d * d;
    #pragma unroll
    for (int o = 16; o > 0; o >>= 1) vq += __shfl_xor(vq, o, 32);
    float rs = rsqrtf(vq * (1.0f / 32.0f) + 1e-5f);   // LN_EPS
    float hn = silu_(d * rs * lng[lane] + lnb[lane]);
    hs[wid][lane] = hn;
    __syncthreads();

    #pragma unroll
    for (int i = 0; i < 4; ++i) {
        int c = lane * 4 + i;
        float o = b2[c];
        const float* w2r = w2 + c * 32;
        #pragma unroll 8
        for (int j = 0; j < 32; ++j) o += hs[wid][j] * w2r[j];
        out[(size_t)t * NCc + c] = sigm_(o);
    }
}

// ------------------------------------------ final fuse + BN + residual, ->NCHW
__global__ __launch_bounds__(256)
void fuse_out_kernel(const float* __restrict__ x,
                     const float* __restrict__ gp, const float* __restrict__ lp,
                     const float* __restrict__ ggl, const float* __restrict__ glg,
                     const float* __restrict__ ns, const float* __restrict__ nsh,
                     float* __restrict__ out)
{
    __shared__ float tile[64 * 132];
    int bh = blockIdx.x;
    int b = bh >> 6, h = bh & 63;
    int tid = threadIdx.x;

    int w = tid >> 2, q = tid & 3;
    size_t t = ((size_t)b << 12) + (h << 6) + w;
    size_t base = t * NCc + q * 32;
    #pragma unroll 4
    for (int j = 0; j < 32; ++j) {
        int c = q * 32 + j;
        float f = gp[base + j] * ggl[base + j] + lp[base + j] * glg[base + j];
        tile[w * 132 + c] = f * ns[c] + nsh[c];
    }
    __syncthreads();

    int c = tid >> 1;
    int w0 = (tid & 1) << 5;
    size_t xoff = (((size_t)b * NCc + c) * NLl) + h * 64 + w0;
    #pragma unroll 4
    for (int j = 0; j < 32; ++j)
        out[xoff + j] = x[xoff + j] + tile[(w0 + j) * 132 + c];
}

// ---------------------------------------------------------------------------
extern "C" void kernel_launch(void* const* d_in, const int* in_sizes, int n_in,
                              void* d_out, int out_size, void* d_ws, size_t ws_size,
                              hipStream_t stream)
{
    (void)in_sizes; (void)n_in; (void)out_size; (void)ws_size;

    // ---- inputs (setup_inputs dict order, depth-first) ----
    const float* x       = (const float*)d_in[0];
    const float* gp_lng  = (const float*)d_in[1];
    const float* gp_lnb  = (const float*)d_in[2];
    const float* ex_w    = (const float*)d_in[3];   // (512,128)
    const float* ex_b    = (const float*)d_in[4];
    const float* proj_w  = (const float*)d_in[5];   // (128,256)
    const float* proj_b  = (const float*)d_in[6];
    const float* gdw_w   = (const float*)d_in[7];   // (256,1,3,3)
    const float* A_log   = (const float*)d_in[8];   // (256,16)
    const float* Dp      = (const float*)d_in[9];
    const float* x_proj  = (const float*)d_in[10];  // (48,256)
    const float* dt_w    = (const float*)d_in[11];  // (256,16)
    const float* dt_b    = (const float*)d_in[12];
    const float* pw_ex   = (const float*)d_in[13];  // (256,128)
    const float* pw_pr   = (const float*)d_in[14];  // (128,256)
    const float* ldw_w   = (const float*)d_in[15];
    const float* bnin_g  = (const float*)d_in[16];
    const float* bnin_b  = (const float*)d_in[17];
    const float* bnin_m  = (const float*)d_in[18];
    const float* bnin_v  = (const float*)d_in[19];
    const float* bn1_g   = (const float*)d_in[20];
    const float* bn1_b   = (const float*)d_in[21];
    const float* bn1_m   = (const float*)d_in[22];
    const float* bn1_v   = (const float*)d_in[23];
    const float* bn2_g   = (const float*)d_in[24];
    const float* bn2_b   = (const float*)d_in[25];
    const float* bn2_m   = (const float*)d_in[26];
    const float* bn2_v   = (const float*)d_in[27];
    const float* bn3_g   = (const float*)d_in[28];
    const float* bn3_b   = (const float*)d_in[29];
    const float* bn3_m   = (const float*)d_in[30];
    const float* bn3_v   = (const float*)d_in[31];
    const float* gl_w1   = (const float*)d_in[32];
    const float* gl_lng  = (const float*)d_in[33];
    const float* gl_lnb  = (const float*)d_in[34];
    const float* gl_w2   = (const float*)d_in[35];
    const float* gl_b2   = (const float*)d_in[36];
    const float* gg_w1   = (const float*)d_in[37];
    const float* gg_lng  = (const float*)d_in[38];
    const float* gg_lnb  = (const float*)d_in[39];
    const float* gg_w2   = (const float*)d_in[40];
    const float* gg_b2   = (const float*)d_in[41];
    const float* nm_g    = (const float*)d_in[42];
    const float* nm_b    = (const float*)d_in[43];
    const float* nm_m    = (const float*)d_in[44];
    const float* nm_v    = (const float*)d_in[45];

    // ---- workspace carve-up ----
    char* ws = (char*)d_ws;
    size_t off = 0;
    auto carve = [&](size_t bytes) -> char* {
        char* p = ws + off;
        off = (off + bytes + 255) & ~(size_t)255;
        return p;
    };
    float* xln   = (float*)carve((size_t)NT * NCc * 4);   // also reused as gp
    float* xbn   = (float*)carve((size_t)NT * NCc * 4);   // also reused as lp
    float* uvb   = (float*)carve((size_t)NT * 512 * 4);
    float* vsb   = (float*)carve((size_t)NT * NCE * 4);
    float* xdblb = (float*)carve((size_t)NT * 48  * 4);
    float* dtb   = (float*)carve((size_t)NT * NCE * 4);
    float* vcomb = (float*)carve((size_t)NT * NCE * 4);
    float* l1s   = (float*)carve((size_t)NT * NCE * 4);
    float* l2b   = (float*)carve((size_t)NT * NCE * 4);
    float* ggl   = (float*)carve((size_t)NT * NCc * 4);   // gating(lp) -> mult gp
    float* glg   = (float*)carve((size_t)NT * NCc * 4);   // gating(gp) -> mult lp
    _Float16* Wex   = (_Float16*)carve(512 * 128 * 2);
    _Float16* Wproj = (_Float16*)carve(128 * 256 * 2);
    _Float16* Wpwex = (_Float16*)carve(256 * 128 * 2);
    _Float16* Wpwpr = (_Float16*)carve(128 * 256 * 2);
    _Float16* Wxp   = (_Float16*)carve(48  * 256 * 2);
    float* bnin_s = (float*)carve(128 * 4);  float* bnin_sh = (float*)carve(128 * 4);
    float* bn1_s  = (float*)carve(256 * 4);  float* bn1_sh  = (float*)carve(256 * 4);
    float* bn2_s  = (float*)carve(256 * 4);  float* bn2_sh  = (float*)carve(256 * 4);
    float* bn3_s  = (float*)carve(128 * 4);  float* bn3_sh  = (float*)carve(128 * 4);
    float* nm_s   = (float*)carve(128 * 4);  float* nm_sh   = (float*)carve(128 * 4);

    float* gpb = xln;   // safe reuse: xln consumed by GEMM B before G runs
    float* lpb = xbn;   // safe reuse: xbn consumed by GEMM H before J runs

    // ---- weight prep ----
    cvt_f16_kernel<<<(65536 + 255) / 256, 256, 0, stream>>>(ex_w,   Wex,   65536);
    cvt_f16_kernel<<<(32768 + 255) / 256, 256, 0, stream>>>(proj_w, Wproj, 32768);
    cvt_f16_kernel<<<(32768 + 255) / 256, 256, 0, stream>>>(pw_ex,  Wpwex, 32768);
    cvt_f16_kernel<<<(32768 + 255) / 256, 256, 0, stream>>>(pw_pr,  Wpwpr, 32768);
    cvt_f16_kernel<<<(12288 + 255) / 256, 256, 0, stream>>>(x_proj, Wxp,   12288);
    bn_affine_kernel<<<1, 256, 0, stream>>>(bnin_g, bnin_b, bnin_m, bnin_v, bnin_s, bnin_sh, 128);
    bn_affine_kernel<<<1, 256, 0, stream>>>(bn1_g,  bn1_b,  bn1_m,  bn1_v,  bn1_s,  bn1_sh,  256);
    bn_affine_kernel<<<1, 256, 0, stream>>>(bn2_g,  bn2_b,  bn2_m,  bn2_v,  bn2_s,  bn2_sh,  256);
    bn_affine_kernel<<<1, 256, 0, stream>>>(bn3_g,  bn3_b,  bn3_m,  bn3_v,  bn3_s,  bn3_sh,  128);
    bn_affine_kernel<<<1, 256, 0, stream>>>(nm_g,   nm_b,   nm_m,   nm_v,   nm_s,   nm_sh,   128);

    // ---- A: LN + BN_in + transpose ----
    lnbn_transpose_kernel<<<NB * NHh, 256, 0, stream>>>(x, gp_lng, gp_lnb, bnin_s, bnin_sh, xln, xbn);

    // ---- global path ----
    wmma_gemm_kernel<4, 0, 1><<<NT / 64, 128, 0, stream>>>(xln, Wex, uvb, 512, nullptr, ex_b);
    dwconv_kernel<<<NT * 64 / 256, 256, 0, stream>>>(uvb + 256, 512, gdw_w, nullptr, nullptr, vsb);
    wmma_gemm_kernel<8, 0, 0><<<NT / 64, 128, 0, stream>>>(vsb, Wxp, xdblb, 48, nullptr, nullptr);
    dtproj_kernel<<<NT * 64 / 256, 256, 0, stream>>>(xdblb, dt_w, dt_b, dtb);
    scan_kernel<<<NB * 8, 32, 0, stream>>>(xdblb, dtb, vsb, uvb, A_log, Dp, vcomb);
    wmma_gemm_kernel<8, 0, 1><<<NT / 64, 128, 0, stream>>>(vcomb, Wproj, gpb, 128, nullptr, proj_b);

    // ---- local path ----
    wmma_gemm_kernel<4, 1, 2><<<NT / 64, 128, 0, stream>>>(xbn, Wpwex, l1s, 256, bn1_s, bn1_sh);
    dwconv_kernel<<<NT * 64 / 256, 256, 0, stream>>>(l1s, 256, ldw_w, bn2_s, bn2_sh, l2b);
    wmma_gemm_kernel<8, 0, 2><<<NT / 64, 128, 0, stream>>>(l2b, Wpwpr, lpb, 128, bn3_s, bn3_sh);

    // ---- gates ----
    gating_kernel<<<NT / 8, 256, 0, stream>>>(lpb, gl_w1, gl_lng, gl_lnb, gl_w2, gl_b2, ggl);
    gating_kernel<<<NT / 8, 256, 0, stream>>>(gpb, gg_w1, gg_lng, gg_lnb, gg_w2, gg_b2, glg);

    // ---- fuse + residual ----
    fuse_out_kernel<<<NB * NHh, 256, 0, stream>>>(x, gpb, lpb, ggl, glg, nm_s, nm_sh, (float*)d_out);
}
